// RecoverySelectiveSSM_19722489823901
// MI455X (gfx1250) — matrix operand order repaired and checked by reading, not compiled, not run on hardware
//
#include <hip/hip_runtime.h>
#include <hip/hip_bf16.h>

#define DEVI __device__ __forceinline__

// ---------------- problem constants ----------------
constexpr int B = 64, S = 1024, DIN = 256, H = 512, Q = 64;
constexpr int NWG = 32;            // scan workgroups (one 16-col slice each)
constexpr int FRAG_BYTES = 1024;   // one 32x16 bf16 WMMA B fragment
constexpr int GATE_BLKS = 18;      // H-row blocks across all 7 gate matrices
constexpr int KT_H = H / 32;       // 16 k-tiles per H block

// ---------------- workspace layout (bytes) ----------------
constexpr size_t OFF_BAR = 0;                                   // global barrier counter
constexpr size_t OFF_SBF = 256;                                 // bf16 state [2][3][B][H]
constexpr size_t SBF_SZ  = (size_t)2 * 3 * B * H * 2;
constexpr size_t OFF_SF  = OFF_SBF + SBF_SZ;                    // f32 state [2][3][B][H]
constexpr size_t SF_SZ   = (size_t)2 * 3 * B * H * 4;
constexpr size_t ZERO_BYTES = OFF_SF + SF_SZ;                   // memset region
constexpr size_t OFF_GW  = ZERO_BYTES;                          // gate frags [32][18][16] x 1KB
constexpr size_t GW_SZ   = (size_t)NWG * GATE_BLKS * KT_H * FRAG_BYTES;
constexpr size_t OFF_WPF = OFF_GW + GW_SZ;                      // Wp frags [32][8] x 1KB
constexpr size_t WPF_SZ  = (size_t)32 * 8 * FRAG_BYTES;
constexpr size_t OFF_WQF = OFF_WPF + WPF_SZ;                    // Wq frags [32][2] x 1KB
constexpr size_t WQF_SZ  = (size_t)32 * 2 * FRAG_BYTES;
constexpr size_t OFF_XP  = OFF_WQF + WQF_SZ;                    // bf16 X [S][B][H]
constexpr size_t XP_SZ   = (size_t)S * B * H * 2;
constexpr size_t OFF_QP  = OFF_XP + XP_SZ;                      // bf16 Q [S][B][H]
constexpr size_t QP_SZ   = XP_SZ;
constexpr size_t OFF_PRE = OFF_QP + QP_SZ;                      // f32 pre-conv [B*S][H]
constexpr size_t PRE_SZ  = (size_t)B * S * H * 4;
constexpr size_t OFF_HID = OFF_PRE + PRE_SZ;                    // f32 hid [B*S][H]

constexpr size_t WLDS_BYTES = (size_t)GATE_BLKS * KT_H * FRAG_BYTES;   // 294912

// ---------------- types / helpers ----------------
typedef __bf16 v16bf __attribute__((ext_vector_type(16)));
typedef float  v8f   __attribute__((ext_vector_type(8)));

union Bf16x16 { v16bf v; uint4 u[2]; };

DEVI v8f wmma_bf16(v16bf a, v16bf b, v8f c) {
  return __builtin_amdgcn_wmma_f32_16x16x32_bf16(false, a, false, b, (short)0, c,
                                                 false, false);
}

// A fragment (16x32, M=row tile, K along columns) gathered from a row-major
// bf16 matrix with row stride `rs` (elements). Per ISA layout: lanes 0-15 hold
// K {k0..k0+7, k0+16..k0+23}; lanes 16-31 hold K {+8} of each half.
DEVI v16bf afrag(const __hip_bfloat16* base, int rs, int row, int k0, int lane) {
  const int hi = (lane >> 4) << 3;
  const __hip_bfloat16* p = base + (size_t)row * rs + k0 + hi;
  Bf16x16 t;
  t.u[0] = *(const uint4*)p;
  t.u[1] = *(const uint4*)(p + 16);
  return t.v;
}

// B fragment (32x16) pre-packed: bytes [0,512) = each lane's first 8 bf16 at
// lane*16, bytes [512,1024) = second 8. Two 128b loads, 2-way-bank-conflict
// optimal in LDS.
DEVI v16bf bfrag(const unsigned char* frag, int lane) {
  Bf16x16 t;
  t.u[0] = *(const uint4*)(frag + lane * 16);
  t.u[1] = *(const uint4*)(frag + 512 + lane * 16);
  return t.v;
}

DEVI float sigf(float x) { return 1.0f / (1.0f + __expf(-x)); }
DEVI float tanhfast(float x) { float e = __expf(-2.0f * x); return (1.0f - e) / (1.0f + e); }

// ---------------- weight packing ----------------
DEVI void pack_one(const float* src, int rowbase, int kt, int lane,
                   unsigned char* frag, int col) {
  __align__(16) __hip_bfloat16 tmp[16];
  const int rb = rowbase + kt * 32 + ((lane >> 4) << 3);
#pragma unroll
  for (int j = 0; j < 16; ++j)
    tmp[j] = __float2bfloat16(src[(size_t)(rb + j + ((j >> 3) << 3)) * H + col]);
  *(uint4*)(frag + lane * 16)       = *(const uint4*)&tmp[0];
  *(uint4*)(frag + 512 + lane * 16) = *(const uint4*)&tmp[8];
}

__global__ __launch_bounds__(256) void pack_weights(
    const float* Wmw, const float* Wmc, const float* Wow, const float* Woc,
    const float* Wrw, const float* Wrc, const float* Wo,
    const float* Wp, const float* Wq, unsigned char* ws) {
  const int gid = blockIdx.x * 256 + threadIdx.x;
  const int NG = NWG * GATE_BLKS * KT_H * 32;     // 294912 gate threads
  const int NP = 32 * 8 * 32;                     // 8192 Wp threads
  const int NQ = 32 * 2 * 32;                     // 2048 Wq threads
  if (gid < NG) {
    const int fid = gid >> 5, lane = gid & 31;
    const int n = fid / (GATE_BLKS * KT_H);
    const int loc = fid % (GATE_BLKS * KT_H);
    const int blk = loc >> 4, kt = loc & 15;
    const float* src; int rb;
    switch (blk) {  // concat order per reference: see gate definitions
      case 0:  src = Wmw; rb = 0;     break;  // mw <- x
      case 1:  src = Wmw; rb = H;     break;  // mw <- q
      case 2:  src = Wmc; rb = 0;     break;  // mc <- x
      case 3:  src = Wmc; rb = H;     break;  // mc <- q
      case 4:  src = Wmc; rb = 2 * H; break;  // mc <- o
      case 5:  src = Wow; rb = 0;     break;  // ow <- q
      case 6:  src = Wow; rb = H;     break;  // ow <- m
      case 7:  src = Woc; rb = 0;     break;  // oc <- q
      case 8:  src = Woc; rb = H;     break;  // oc <- m
      case 9:  src = Woc; rb = 2 * H; break;  // oc <- r
      case 10: src = Wrw; rb = 0;     break;  // rw <- q
      case 11: src = Wrw; rb = H;     break;  // rw <- o
      case 12: src = Wrc; rb = 0;     break;  // rc <- x
      case 13: src = Wrc; rb = H;     break;  // rc <- m
      case 14: src = Wrc; rb = 2 * H; break;  // rc <- o
      case 15: src = Wo;  rb = 0;     break;  // out <- nm
      case 16: src = Wo;  rb = H;     break;  // out <- no
      default: src = Wo;  rb = 2 * H; break;  // out <- nr
    }
    pack_one(src, rb, kt, lane, ws + OFF_GW + (size_t)fid * FRAG_BYTES,
             n * 16 + (lane & 15));
  } else if (gid < NG + NP) {
    const int g2 = gid - NG, fid = g2 >> 5, lane = g2 & 31;
    const int n = fid >> 3, kt = fid & 7;
    pack_one(Wp, 0, kt, lane, ws + OFF_WPF + (size_t)fid * FRAG_BYTES,
             n * 16 + (lane & 15));
  } else if (gid < NG + NP + NQ) {
    const int g3 = gid - NG - NP, fid = g3 >> 5, lane = g3 & 31;
    const int n = fid >> 1, kt = fid & 1;
    pack_one(Wq, 0, kt, lane, ws + OFF_WQF + (size_t)fid * FRAG_BYTES,
             n * 16 + (lane & 15));
  }
}

// ---------------- LayerNorm + input projection (WMMA) ----------------
__global__ __launch_bounds__(256) void prep_proj(
    const float* __restrict__ x, const float* __restrict__ lg,
    const float* __restrict__ lb, const float* __restrict__ bp,
    const unsigned char* __restrict__ WpF, float* __restrict__ Pre) {
  __shared__ __hip_bfloat16 xn[16 * DIN];
  const int tid = threadIdx.x, wave = tid >> 5, lane = tid & 31;
  const int r0 = blockIdx.x * 16;
  // each wave LayerNorms 2 of the 16 rows (256 elems -> 8 per lane, wave32 shuffles)
  for (int rr = 0; rr < 2; ++rr) {
    const int rl = wave * 2 + rr;
    const float* xr = x + (size_t)(r0 + rl) * DIN + lane * 8;
    float v[8]; float s0 = 0.f;
#pragma unroll
    for (int j = 0; j < 8; ++j) { v[j] = xr[j]; s0 += v[j]; }
#pragma unroll
    for (int o = 16; o > 0; o >>= 1) s0 += __shfl_xor(s0, o, 32);
    const float mu = s0 * (1.0f / DIN);
    float s1 = 0.f;
#pragma unroll
    for (int j = 0; j < 8; ++j) { float d = v[j] - mu; s1 += d * d; }
#pragma unroll
    for (int o = 16; o > 0; o >>= 1) s1 += __shfl_xor(s1, o, 32);
    const float inv = rsqrtf(s1 * (1.0f / DIN) + 1e-5f);
#pragma unroll
    for (int j = 0; j < 8; ++j) {
      const int c = lane * 8 + j;
      xn[rl * DIN + c] = __float2bfloat16((v[j] - mu) * inv * lg[c] + lb[c]);
    }
  }
  __syncthreads();
  const int rhi = (lane >> 4) << 3;
  for (int n = wave; n < H / 16; n += 8) {
    v8f acc = {};
#pragma unroll
    for (int kt = 0; kt < DIN / 32; ++kt) {
      v16bf a  = afrag(xn, DIN, lane & 15, kt * 32, lane);
      v16bf bb = bfrag(WpF + (size_t)(n * (DIN / 32) + kt) * FRAG_BYTES, lane);
      acc = wmma_bf16(a, bb, acc);
    }
    const int colg = n * 16 + (lane & 15);
    const float bias = bp[colg];
#pragma unroll
    for (int r = 0; r < 8; ++r)
      Pre[(size_t)(r0 + r + rhi) * H + colg] = acc[r] + bias;
  }
}

// ---------------- quality projection (WMMA) -> bf16 [S][B][H] ----------------
__global__ __launch_bounds__(256) void prep_qe(
    const float* __restrict__ q, const float* __restrict__ bq,
    const unsigned char* __restrict__ WqF, __hip_bfloat16* __restrict__ Qp) {
  __shared__ __hip_bfloat16 ql[16 * Q];
  const int tid = threadIdx.x, wave = tid >> 5, lane = tid & 31;
  const int r0 = blockIdx.x * 16;
#pragma unroll
  for (int i = 0; i < 4; ++i) {
    const int idx = tid * 4 + i;
    const int rl = idx >> 6, c = idx & 63;
    ql[idx] = __float2bfloat16(q[(size_t)(r0 + rl) * Q + c]);
  }
  __syncthreads();
  const int rhi = (lane >> 4) << 3;
  for (int n = wave; n < H / 16; n += 8) {
    v8f acc = {};
#pragma unroll
    for (int kt = 0; kt < Q / 32; ++kt) {
      v16bf a  = afrag(ql, Q, lane & 15, kt * 32, lane);
      v16bf bb = bfrag(WqF + (size_t)(n * (Q / 32) + kt) * FRAG_BYTES, lane);
      acc = wmma_bf16(a, bb, acc);
    }
    const int colg = n * 16 + (lane & 15);
    const float bias = bq[colg];
#pragma unroll
    for (int r = 0; r < 8; ++r) {
      const int row = r0 + r + rhi;
      const int bb2 = row >> 10, ss = row & (S - 1);
      Qp[((size_t)ss * B + bb2) * H + colg] = __float2bfloat16(acc[r] + bias);
    }
  }
}

// ---------------- depthwise conv3 + residual source + bf16 pack ----------------
__global__ __launch_bounds__(256) void conv_pack(
    const float* __restrict__ Pre, const float* __restrict__ Wc,
    const float* __restrict__ bc, float* __restrict__ Hid,
    __hip_bfloat16* __restrict__ Xp) {
  const size_t idx = (size_t)blockIdx.x * 256 + threadIdx.x;
  if (idx >= (size_t)B * S * H) return;
  const int h = (int)(idx & (H - 1));
  const size_t r = idx >> 9;                // b*S + s
  const int s = (int)(r & (S - 1));
  const int b = (int)(r >> 10);
  float acc = bc[h];
  const float w0 = Wc[h * 3 + 0], w1 = Wc[h * 3 + 1], w2 = Wc[h * 3 + 2];
  if (s > 0)     acc += Pre[(r - 1) * H + h] * w0;
  acc += Pre[r * H + h] * w1;
  if (s < S - 1) acc += Pre[(r + 1) * H + h] * w2;
  Hid[r * H + h] = acc;
  Xp[((size_t)s * B + b) * H + h] = __float2bfloat16(acc);
}

// ---------------- persistent recurrent scan ----------------
#define WF(blk, kt) (wlds + (((blk) * KT_H + (kt)) * FRAG_BYTES))

__global__ __launch_bounds__(256, 1) void scan_kernel(
    const unsigned char* __restrict__ GW,
    const float* __restrict__ bmw, const float* __restrict__ bmc,
    const float* __restrict__ bow, const float* __restrict__ boc,
    const float* __restrict__ brw, const float* __restrict__ brc,
    const float* __restrict__ bo,
    const int* __restrict__ mask,
    const __hip_bfloat16* __restrict__ Xp, const __hip_bfloat16* __restrict__ Qp,
    const float* __restrict__ Hid,
    __hip_bfloat16* __restrict__ Sbf, float* __restrict__ Sf,
    int* __restrict__ bar, float* __restrict__ outp) {
  extern __shared__ unsigned char smem[];
  unsigned char* wlds = smem;                               // 288 KB gate weights
  float* gbuf = (float*)(smem + WLDS_BYTES);                // 6 x [64][16] gate pre-acts
  float* blds = gbuf + 6 * B * 16;                          // 7 x 16 biases
  const int tid = threadIdx.x, wave = tid >> 5, lane = tid & 31;
  const int wg = blockIdx.x, h0 = wg * 16;

  {  // stage this slice's gate weights (LDS-resident for the whole scan)
    const uint4* src = (const uint4*)(GW + (size_t)wg * WLDS_BYTES);
    uint4* dst = (uint4*)wlds;
    for (int i = tid; i < (int)(WLDS_BYTES / 16); i += 256) dst[i] = src[i];
    if (tid < 7 * 16) {
      const float* bp7[7] = {bmw, bmc, bow, boc, brw, brc, bo};
      blds[tid] = bp7[tid >> 4][h0 + (tid & 15)];
    }
  }
  __syncthreads();

  const int mt = wave & 3, grp = wave >> 2, b0 = mt * 16;
  const int arow = b0 + (lane & 15);
  const int col = lane & 15;
  const int rhi = (lane >> 4) << 3;

  for (int s = 0; s <= S; ++s) {
    const int cur = s & 1;
    const __hip_bfloat16* Sc = Sbf + (size_t)cur * 3 * B * H;
    const __hip_bfloat16* Xs = Xp + (size_t)s * B * H;
    const __hip_bfloat16* Qs = Qp + (size_t)s * B * H;

    if (grp == 0) {
      // sigmoid gates (K=2H each) + pipelined Wo GEMM for step s-1 (K=3H)
      v8f aMW = {}, aOW = {}, aRW = {}, aOUT = {};
      for (int kt = 0; kt < KT_H; ++kt) {
        const int k0 = kt * 32;
        v16bf ma = afrag(Sc,             H, arow, k0, lane);
        v16bf oa = afrag(Sc + B * H,     H, arow, k0, lane);
        v16bf ra = afrag(Sc + 2 * B * H, H, arow, k0, lane);
        if (s < S) {
          v16bf xa = afrag(Xs, H, arow, k0, lane);
          v16bf qa = afrag(Qs, H, arow, k0, lane);
          aMW = wmma_bf16(xa, bfrag(WF(0, kt), lane), aMW);
          aMW = wmma_bf16(qa, bfrag(WF(1, kt), lane), aMW);
          aOW = wmma_bf16(qa, bfrag(WF(5, kt), lane), aOW);
          aOW = wmma_bf16(ma, bfrag(WF(6, kt), lane), aOW);
          aRW = wmma_bf16(qa, bfrag(WF(10, kt), lane), aRW);
          aRW = wmma_bf16(oa, bfrag(WF(11, kt), lane), aRW);
        }
        if (s > 0) {
          aOUT = wmma_bf16(ma, bfrag(WF(15, kt), lane), aOUT);
          aOUT = wmma_bf16(oa, bfrag(WF(16, kt), lane), aOUT);
          aOUT = wmma_bf16(ra, bfrag(WF(17, kt), lane), aOUT);
        }
      }
      if (s > 0) {
        const float biasO = blds[6 * 16 + col];
#pragma unroll
        for (int r = 0; r < 8; ++r) {
          const int bb = b0 + r + rhi;
          const size_t row = ((size_t)bb << 10) + (s - 1);
          outp[row * H + h0 + col] = aOUT[r] + biasO + Hid[row * H + h0 + col];
        }
      }
      if (s < S) {
#pragma unroll
        for (int r = 0; r < 8; ++r) {
          const int rw_ = (b0 + r + rhi) * 16 + col;
          gbuf[0 * 1024 + rw_] = aMW[r];
          gbuf[2 * 1024 + rw_] = aOW[r];
          gbuf[4 * 1024 + rw_] = aRW[r];
        }
      }
    } else if (s < S) {
      // tanh candidates (K=3H each)
      v8f aMC = {}, aOC = {}, aRC = {};
      for (int kt = 0; kt < KT_H; ++kt) {
        const int k0 = kt * 32;
        v16bf ma = afrag(Sc,             H, arow, k0, lane);
        v16bf oa = afrag(Sc + B * H,     H, arow, k0, lane);
        v16bf ra = afrag(Sc + 2 * B * H, H, arow, k0, lane);
        v16bf xa = afrag(Xs, H, arow, k0, lane);
        v16bf qa = afrag(Qs, H, arow, k0, lane);
        aMC = wmma_bf16(xa, bfrag(WF(2, kt), lane), aMC);
        aMC = wmma_bf16(qa, bfrag(WF(3, kt), lane), aMC);
        aMC = wmma_bf16(oa, bfrag(WF(4, kt), lane), aMC);
        aOC = wmma_bf16(qa, bfrag(WF(7, kt), lane), aOC);
        aOC = wmma_bf16(ma, bfrag(WF(8, kt), lane), aOC);
        aOC = wmma_bf16(ra, bfrag(WF(9, kt), lane), aOC);
        aRC = wmma_bf16(xa, bfrag(WF(12, kt), lane), aRC);
        aRC = wmma_bf16(ma, bfrag(WF(13, kt), lane), aRC);
        aRC = wmma_bf16(oa, bfrag(WF(14, kt), lane), aRC);
      }
#pragma unroll
      for (int r = 0; r < 8; ++r) {
        const int rw_ = (b0 + r + rhi) * 16 + col;
        gbuf[1 * 1024 + rw_] = aMC[r];
        gbuf[3 * 1024 + rw_] = aOC[r];
        gbuf[5 * 1024 + rw_] = aRC[r];
      }
    }

    if (s < S) {
      __syncthreads();
      // fused gate nonlinearities + masked state update for our 16 columns
      const float* SfC = Sf + (size_t)cur * 3 * B * H;
      float* SfN = Sf + (size_t)(cur ^ 1) * 3 * B * H;
      __hip_bfloat16* SbN = Sbf + (size_t)(cur ^ 1) * 3 * B * H;
#pragma unroll
      for (int i = 0; i < 4; ++i) {
        const int idx = tid + i * 256;          // 64 rows x 16 cols
        const int bb = idx >> 4, c = idx & 15;
        const size_t off = (size_t)bb * H + h0 + c;
        const float mw = sigf(gbuf[0 * 1024 + idx] + blds[0 * 16 + c]);
        const float mc = tanhfast(gbuf[1 * 1024 + idx] + blds[1 * 16 + c]);
        const float ow = sigf(gbuf[2 * 1024 + idx] + blds[2 * 16 + c]);
        const float oc = tanhfast(gbuf[3 * 1024 + idx] + blds[3 * 16 + c]);
        const float rw = sigf(gbuf[4 * 1024 + idx] + blds[4 * 16 + c]);
        const float rc = tanhfast(gbuf[5 * 1024 + idx] + blds[5 * 16 + c]);
        const float m0 = SfC[0 * B * H + off];
        const float o0 = SfC[1 * B * H + off];
        const float r0s = SfC[2 * B * H + off];
        const bool v = mask[((size_t)bb << 10) + s] > 0;
        const float nm = v ? (1.f - mw) * m0 + mw * mc : m0;
        const float no = v ? (1.f - ow) * o0 + ow * oc : o0;
        const float nr = v ? (1.f - rw) * r0s + rw * rc : r0s;
        SfN[0 * B * H + off] = nm;
        SfN[1 * B * H + off] = no;
        SfN[2 * B * H + off] = nr;
        SbN[0 * B * H + off] = __float2bfloat16(nm);
        SbN[1 * B * H + off] = __float2bfloat16(no);
        SbN[2 * B * H + off] = __float2bfloat16(nr);
      }
      __threadfence();            // release our state slice device-wide
      __syncthreads();
      if (tid == 0) {             // device-scope counter barrier across 32 WGs
        (void)__hip_atomic_fetch_add(bar, 1, __ATOMIC_ACQ_REL,
                                     __HIP_MEMORY_SCOPE_AGENT);
        const int target = NWG * (s + 1);
        while (__hip_atomic_load(bar, __ATOMIC_ACQUIRE,
                                 __HIP_MEMORY_SCOPE_AGENT) < target)
          __builtin_amdgcn_s_sleep(2);
      }
      __syncthreads();
      __threadfence();            // acquire: drop stale cached state lines
    }
  }

  // final carries (state after step S-1 lives in buffer parity S&1 == 0)
  {
    const size_t base = (size_t)B * S * H;
#pragma unroll
    for (int i = 0; i < 12; ++i) {
      const int idx = tid + i * 256;           // 3 * 64 * 16 = 3072
      if (idx < 3 * B * 16) {
        const int st = idx >> 10, rem = idx & 1023, bb = rem >> 4, c = rem & 15;
        const size_t off = (size_t)st * B * H + (size_t)bb * H + h0 + c;
        outp[base + off] = Sf[off];
      }
    }
  }
}

// ---------------- host entry ----------------
extern "C" void kernel_launch(void* const* d_in, const int* in_sizes, int n_in,
                              void* d_out, int out_size, void* d_ws, size_t ws_size,
                              hipStream_t stream) {
  (void)in_sizes; (void)n_in; (void)out_size; (void)ws_size;
  const float* seq  = (const float*)d_in[0];
  const float* qual = (const float*)d_in[1];
  const int*   mask = (const int*)d_in[2];
  const float* ln_g = (const float*)d_in[3];
  const float* ln_b = (const float*)d_in[4];
  const float* Wp   = (const float*)d_in[5];
  const float* bp   = (const float*)d_in[6];
  const float* Wc   = (const float*)d_in[7];
  const float* bc   = (const float*)d_in[8];
  const float* Wq   = (const float*)d_in[9];
  const float* bq   = (const float*)d_in[10];
  const float* Wmw  = (const float*)d_in[11];
  const float* bmw  = (const float*)d_in[12];
  const float* Wmc  = (const float*)d_in[13];
  const float* bmc  = (const float*)d_in[14];
  const float* Wow  = (const float*)d_in[15];
  const float* bow  = (const float*)d_in[16];
  const float* Woc  = (const float*)d_in[17];
  const float* boc  = (const float*)d_in[18];
  const float* Wrw  = (const float*)d_in[19];
  const float* brw  = (const float*)d_in[20];
  const float* Wrc  = (const float*)d_in[21];
  const float* brc  = (const float*)d_in[22];
  const float* Wo   = (const float*)d_in[23];
  const float* bo   = (const float*)d_in[24];
  unsigned char* ws = (unsigned char*)d_ws;
  float* outp = (float*)d_out;

  // zero barrier counter + both state double-buffers (graph-capture safe)
  hipMemsetAsync(ws, 0, ZERO_BYTES, stream);

  pack_weights<<<(NWG * GATE_BLKS * KT_H * 32 + 32 * 8 * 32 + 32 * 2 * 32) / 256,
                 256, 0, stream>>>(Wmw, Wmc, Wow, Woc, Wrw, Wrc, Wo, Wp, Wq, ws);

  prep_proj<<<(B * S) / 16, 256, 0, stream>>>(
      seq, ln_g, ln_b, bp, ws + OFF_WPF, (float*)(ws + OFF_PRE));

  prep_qe<<<(B * S) / 16, 256, 0, stream>>>(
      qual, bq, ws + OFF_WQF, (__hip_bfloat16*)(ws + OFF_QP));

  conv_pack<<<(B * S * H) / 256, 256, 0, stream>>>(
      (const float*)(ws + OFF_PRE), Wc, bc, (float*)(ws + OFF_HID),
      (__hip_bfloat16*)(ws + OFF_XP));

  const size_t smem = WLDS_BYTES + (size_t)6 * B * 16 * 4 + 7 * 16 * 4;  // 319936 B
  scan_kernel<<<NWG, 256, smem, stream>>>(
      ws + OFF_GW, bmw, bmc, bow, boc, brw, brc, bo, mask,
      (const __hip_bfloat16*)(ws + OFF_XP), (const __hip_bfloat16*)(ws + OFF_QP),
      (const float*)(ws + OFF_HID),
      (__hip_bfloat16*)(ws + OFF_SBF), (float*)(ws + OFF_SF),
      (int*)(ws + OFF_BAR), outp);
}
